// BahdanauAttention_8787503087869
// MI455X (gfx1250) — matrix operand order repaired and checked by reading, not compiled, not run on hardware
//
#include <hip/hip_runtime.h>
#include <hip/hip_bf16.h>

typedef __attribute__((ext_vector_type(16))) __bf16 v16bf;
typedef __attribute__((ext_vector_type(8)))  float  v8f;
typedef __attribute__((ext_vector_type(4)))  float  f4v;
typedef __attribute__((ext_vector_type(4)))  __bf16 bf4;

#define BB    32
#define TX    1024
#define EE    1024
#define DD    1024
#define AA    1024
#define WROW  2048          // W_w row length = enc+dec
#define NEG_INF_F (-1.0e9f)

// packed f32 -> bf16 conversion: lowers to v_cvt_pk_bf16_f32 (RNE)
__device__ __forceinline__ void store4bf(unsigned short* p, float4 v) {
    f4v f = {v.x, v.y, v.z, v.w};
    bf4 b = __builtin_convertvector(f, bf4);
    *(bf4*)p = b;
}

// branch-free tanh: 2 trans ops (exp, rcp) + few VALU; ~1e-7 rel err (GEMM err is ~1e-3)
__device__ __forceinline__ float tanh_fast(float x) {
    float xc = fminf(fmaxf(x, -15.f), 15.f);
    float e  = __expf(2.f * xc);
    return 1.f - 2.f * __builtin_amdgcn_rcpf(e + 1.f);
}

// CDNA5 async global->LDS copy (16 B per lane), tracked by ASYNCcnt
__device__ __forceinline__ void async_b128(unsigned ldsAddr, const void* gptr) {
    asm volatile("global_load_async_to_lds_b128 %0, %1, off"
                 :: "v"(ldsAddr), "v"(gptr) : "memory");
}
__device__ __forceinline__ void wait_asynccnt0() {
    asm volatile("s_wait_asynccnt 0" ::: "memory");
}

union FragU { float4 f[2]; v16bf v; };

// ---------------------------------------------------------------------------
// Kernel 0: one-time W_enc f32 -> bf16   Wbf[a][e] = bf16(W_w[a][DD+e])
// ---------------------------------------------------------------------------
__global__ void k0_convW(const float* __restrict__ Ww,
                         unsigned short* __restrict__ Wbf) {
    const int idx = blockIdx.x * 256 + threadIdx.x;  // 0 .. 131071
    const int a   = idx >> 7;                        // 128 chunks of 8 per row
    const int e8  = (idx & 127) * 8;
    const float* src = Ww + (size_t)a * WROW + DD + e8;
    float4 v0 = *(const float4*)src;
    float4 v1 = *(const float4*)(src + 4);
    unsigned short* dst = Wbf + (size_t)a * EE + e8;
    store4bf(dst, v0);
    store4bf(dst + 4, v1);
}

// ---------------------------------------------------------------------------
// Kernel 1: base[b][a] = sum_d dec[b][d] * W_w[a][d] + W_b[a]
// ---------------------------------------------------------------------------
__global__ void k1_base(const float* __restrict__ dec,
                        const float* __restrict__ Ww,
                        const float* __restrict__ Wb,
                        float* __restrict__ base) {
    const int b = blockIdx.x >> 2;
    const int a = ((blockIdx.x & 3) << 8) + threadIdx.x;
    __shared__ float decS[DD];
    #pragma unroll
    for (int j = 0; j < 4; ++j)
        decS[threadIdx.x + 256 * j] = dec[(size_t)b * DD + threadIdx.x + 256 * j];
    __syncthreads();

    const float4* Wr = (const float4*)(Ww + (size_t)a * WROW);
    float acc = 0.f;
    #pragma unroll 4
    for (int d4 = 0; d4 < DD / 4; ++d4) {
        float4 w4 = Wr[d4];
        const float* ds = &decS[d4 * 4];
        acc += w4.x * ds[0] + w4.y * ds[1] + w4.z * ds[2] + w4.w * ds[3];
    }
    base[(size_t)b * AA + a] = acc + Wb[a];
}

// ---------------------------------------------------------------------------
// Kernel 2: fused  scores[b][t] = sum_a tanh(base[b][a] + enc[b][t]·W_enc[a]) * v[a]
// grid: (8 t-tiles, 32 b) x 256 threads (8 waves).
// Double-buffered LDS; B via async global->LDS (bf16), A converted in flight.
// One barrier per K-step; DS-read/WMMA interleave enforced via sched_group_barrier.
// ---------------------------------------------------------------------------
__global__ void __launch_bounds__(256)
k2_scores(const float*          __restrict__ enc,
          const unsigned short* __restrict__ Wbf,
          const float*          __restrict__ base,
          const float*          __restrict__ vw,
          float*                __restrict__ scores) {
    const int b     = blockIdx.y;
    const int tBase = blockIdx.x * 128;
    const int tid   = threadIdx.x;
    const int wave  = tid >> 5;
    const int lane  = tid & 31;
    const int h     = lane >> 4;     // lane half
    const int ln    = lane & 15;

    // pad rows to 40 bf16 (80 B) -> conflict-free b128 fragment gathers
    __shared__ unsigned short As[2][128 * 40];   // A tile: 128 t-rows x 32 k
    __shared__ unsigned short Bs[2][128 * 40];   // B tile (n-major): 128 a-rows x 32 k
    __shared__ float scorePart[8][128];
    __shared__ float scoreAcc[128];

    if (tid < 128) scoreAcc[tid] = 0.f;

    // per-thread staging geometry
    const int rA[4]  = { (tid + 0) >> 3, (tid + 256) >> 3, (tid + 512) >> 3, (tid + 768) >> 3 };
    const int cA     = (tid & 7) * 4;                 // float col within row
    const int rB[2]  = { (tid + 0) >> 2, (tid + 256) >> 2 };
    const int cB     = (tid & 3) * 8;                 // bf16 col within row

    // LDS byte addresses (low 32 bits of flat addr == LDS offset)
    const unsigned bLds[2] = {
        (unsigned)(size_t)&Bs[0][rB[0] * 40 + cB],
        (unsigned)(size_t)&Bs[0][rB[1] * 40 + cB] };
    const unsigned bufStride = (unsigned)(128 * 40 * sizeof(unsigned short));

    const float* encB = enc + ((size_t)b * TX + tBase) * EE;

    for (int chunk = 0; chunk < 8; ++chunk) {
        const int aBase = chunk * 128;
        v8f c[8];
        #pragma unroll
        for (int i = 0; i < 8; ++i) c[i] = v8f{};

        float4 aReg[4];

        // ---- prologue: stage k0 = 0 into buffer 0 ----
        #pragma unroll
        for (int j = 0; j < 2; ++j)
            async_b128(bLds[j], Wbf + (size_t)(aBase + rB[j]) * EE + cB);
        #pragma unroll
        for (int j = 0; j < 4; ++j)
            aReg[j] = *(const float4*)(encB + (size_t)rA[j] * EE + cA);
        #pragma unroll
        for (int j = 0; j < 4; ++j) store4bf(&As[0][rA[j] * 40 + cA], aReg[j]);
        wait_asynccnt0();
        __syncthreads();

        int cur = 0;
        for (int k0 = 0; k0 < EE; k0 += 32) {
            const bool more = (k0 + 32) < EE;
            const int  nxt  = cur ^ 1;
            // ---- issue next tile's loads up front (overlap with compute) ----
            if (more) {
                #pragma unroll
                for (int j = 0; j < 2; ++j)
                    async_b128(bLds[j] + (unsigned)nxt * bufStride,
                               Wbf + (size_t)(aBase + rB[j]) * EE + k0 + 32 + cB);
                #pragma unroll
                for (int j = 0; j < 4; ++j)
                    aReg[j] = *(const float4*)(encB + (size_t)rA[j] * EE + k0 + 32 + cA);
            }

            // ---- compute on current buffer (2-deep pipelined fragment loads) ----
            {
                const char* abase_p = (const char*)&As[cur][0];
                const char* bbase_p = (const char*)&Bs[cur][0];
                auto loadA = [&](int i) {
                    FragU u;
                    const char* ap = abase_p + (i * 16 + ln) * 80 + 16 * h;
                    u.f[0] = *(const float4*)ap;         // K = 8h .. 8h+7
                    u.f[1] = *(const float4*)(ap + 32);  // K = 16+8h .. 23+8h
                    return u;
                };
                FragU bU;
                const char* bp = bbase_p + (wave * 16 + ln) * 80 + 16 * h;
                bU.f[0] = *(const float4*)bp;
                bU.f[1] = *(const float4*)(bp + 32);

                FragU a0 = loadA(0);
                FragU a1 = loadA(1);
                c[0] = __builtin_amdgcn_wmma_f32_16x16x32_bf16(false, a0.v, false, bU.v, (short)0, c[0], false, false);
                FragU a2 = loadA(2);
                c[1] = __builtin_amdgcn_wmma_f32_16x16x32_bf16(false, a1.v, false, bU.v, (short)0, c[1], false, false);
                FragU a3 = loadA(3);
                c[2] = __builtin_amdgcn_wmma_f32_16x16x32_bf16(false, a2.v, false, bU.v, (short)0, c[2], false, false);
                FragU a4 = loadA(4);
                c[3] = __builtin_amdgcn_wmma_f32_16x16x32_bf16(false, a3.v, false, bU.v, (short)0, c[3], false, false);
                FragU a5 = loadA(5);
                c[4] = __builtin_amdgcn_wmma_f32_16x16x32_bf16(false, a4.v, false, bU.v, (short)0, c[4], false, false);
                FragU a6 = loadA(6);
                c[5] = __builtin_amdgcn_wmma_f32_16x16x32_bf16(false, a5.v, false, bU.v, (short)0, c[5], false, false);
                FragU a7 = loadA(7);
                c[6] = __builtin_amdgcn_wmma_f32_16x16x32_bf16(false, a6.v, false, bU.v, (short)0, c[6], false, false);
                c[7] = __builtin_amdgcn_wmma_f32_16x16x32_bf16(false, a7.v, false, bU.v, (short)0, c[7], false, false);

                // enforce DS-read / WMMA interleave: 6 reads, then {1 wmma, 2 reads}x6, 2 wmma
                __builtin_amdgcn_sched_group_barrier(0x100, 6, 0);   // DS read x6 (B, A0, A1)
                #pragma unroll
                for (int g = 0; g < 6; ++g) {
                    __builtin_amdgcn_sched_group_barrier(0x008, 1, 0); // 1 WMMA
                    __builtin_amdgcn_sched_group_barrier(0x100, 2, 0); // next A frag
                }
                __builtin_amdgcn_sched_group_barrier(0x008, 2, 0);     // last 2 WMMA
            }

            // ---- publish next buffer: A stores + async completion, 1 barrier ----
            if (more) {
                #pragma unroll
                for (int j = 0; j < 4; ++j) store4bf(&As[nxt][rA[j] * 40 + cA], aReg[j]);
            }
            wait_asynccnt0();
            __syncthreads();
            cur = nxt;
        }

        // ---- epilogue: tanh, scale by v, reduce over this wave's 16 a-cols ----
        const int aIdx = aBase + wave * 16 + ln;
        const float baseV = base[(size_t)b * AA + aIdx];
        const float vV    = vw[aIdx];
        #pragma unroll
        for (int i = 0; i < 8; ++i) {
            #pragma unroll
            for (int r = 0; r < 8; ++r) {
                float val = tanh_fast(baseV + c[i][r]) * vV;
                val += __shfl_xor(val, 1, 16);
                val += __shfl_xor(val, 2, 16);
                val += __shfl_xor(val, 4, 16);
                val += __shfl_xor(val, 8, 16);
                if (ln == 0)                       // lanes 0 and 16
                    scorePart[wave][i * 16 + r + 8 * h] = val;
            }
        }
        __syncthreads();
        if (tid < 128) {                           // fixed-order -> deterministic
            float s = 0.f;
            #pragma unroll
            for (int w = 0; w < 8; ++w) s += scorePart[w][tid];
            scoreAcc[tid] += s;
        }
        __syncthreads();
    }
    if (tid < 128) scores[(size_t)b * TX + tBase + tid] = scoreAcc[tid];
}

// ---------------------------------------------------------------------------
// Kernel 3: masked softmax over Tx, then context[b][e] = sum_t alpha * enc.
// d_out: [0,32768) context, [32768,65536) alpha.
// ---------------------------------------------------------------------------
__global__ void __launch_bounds__(256)
k3_softmax_ctx(const float* __restrict__ scores,
               const int*   __restrict__ mask,
               const float* __restrict__ enc,
               float* __restrict__ out) {
    const int b   = blockIdx.x;
    const int tid = threadIdx.x;
    __shared__ float alphaS[TX];
    __shared__ float red[256];

    float mx = NEG_INF_F;
    #pragma unroll
    for (int j = 0; j < 4; ++j) {
        const int t = tid + 256 * j;
        float s = (mask[(size_t)b * TX + t] == 0) ? NEG_INF_F
                                                  : scores[(size_t)b * TX + t];
        alphaS[t] = s;
        mx = fmaxf(mx, s);
    }
    red[tid] = mx; __syncthreads();
    for (int s = 128; s > 0; s >>= 1) {
        if (tid < s) red[tid] = fmaxf(red[tid], red[tid + s]);
        __syncthreads();
    }
    mx = red[0]; __syncthreads();

    float lsum = 0.f;
    #pragma unroll
    for (int j = 0; j < 4; ++j) {
        const int t = tid + 256 * j;
        float e = __expf(alphaS[t] - mx);
        alphaS[t] = e;
        lsum += e;
    }
    red[tid] = lsum; __syncthreads();
    for (int s = 128; s > 0; s >>= 1) {
        if (tid < s) red[tid] += red[tid + s];
        __syncthreads();
    }
    const float inv = 1.f / red[0];
    __syncthreads();

    #pragma unroll
    for (int j = 0; j < 4; ++j) {
        const int t = tid + 256 * j;
        float a = alphaS[t] * inv;
        alphaS[t] = a;
        out[(size_t)BB * EE + (size_t)b * TX + t] = a;   // alpha region
    }
    __syncthreads();

    // context: each thread owns 4 consecutive e columns (float4)
    const float4* encV = (const float4*)(enc + (size_t)b * TX * EE);
    float4 acc = make_float4(0.f, 0.f, 0.f, 0.f);
    for (int t4 = 0; t4 < TX / 4; ++t4) {
        float4 al = *(const float4*)&alphaS[t4 * 4];
        #pragma unroll
        for (int u = 0; u < 4; ++u) {
            const float a = (&al.x)[u];
            float4 x = encV[(size_t)(t4 * 4 + u) * (EE / 4) + tid];
            acc.x += a * x.x; acc.y += a * x.y; acc.z += a * x.z; acc.w += a * x.w;
        }
    }
    *(float4*)(out + (size_t)b * EE + tid * 4) = acc;
}

// ---------------------------------------------------------------------------
extern "C" void kernel_launch(void* const* d_in, const int* in_sizes, int n_in,
                              void* d_out, int out_size, void* d_ws, size_t ws_size,
                              hipStream_t stream) {
    const float* dec  = (const float*)d_in[0];   // (32,1024)
    const float* enc  = (const float*)d_in[1];   // (32,1024,1024)
    const int*   mask = (const int*)  d_in[2];   // (32,1024)
    const float* Ww   = (const float*)d_in[3];   // (1024,2048)
    const float* Wb   = (const float*)d_in[4];   // (1024,)
    const float* vw   = (const float*)d_in[5];   // (1,1024)

    float* base   = (float*)d_ws;                          // 32*1024 f32
    float* scores = base + (size_t)BB * AA;                // 32*1024 f32
    unsigned short* Wbf = (unsigned short*)(scores + (size_t)BB * TX); // 1024*1024 bf16

    k0_convW<<<512, 256, 0, stream>>>(Ww, Wbf);
    k1_base<<<128, 256, 0, stream>>>(dec, Ww, Wb, base);

    dim3 g2(8, BB);
    k2_scores<<<g2, 256, 0, stream>>>(enc, Wbf, base, vw, scores);

    k3_softmax_ctx<<<BB, 256, 0, stream>>>(scores, mask, enc, (float*)d_out);
}
